// Mem_Net_76012331205101
// MI455X (gfx1250) — matrix operand order repaired and checked by reading, not compiled, not run on hardware
//
#include <hip/hip_runtime.h>
#include <hip/hip_bf16.h>
#include <math.h>

typedef __attribute__((ext_vector_type(16))) _Float16 v16h;
typedef __attribute__((ext_vector_type(8)))  _Float16 v8h;
typedef __attribute__((ext_vector_type(8)))  float    v8f;

#define WMMA_F16(a,b,c) __builtin_amdgcn_wmma_f32_16x16x32_f16(false,(a),false,(b),(short)0,(c),false,false)

// ---------------------------------------------------------------------------
// WMMA fragment helpers (contiguous k order, k = 16*(lane>>4) + 0..15, wave32)
// ---------------------------------------------------------------------------
__device__ inline v16h make_frag(v8h lo, v8h hi) {
    v16h r;
#pragma unroll
    for (int i = 0; i < 8; ++i) { r[i] = lo[i]; r[i + 8] = hi[i]; }
    return r;
}

// A matrix (MxK, 16x32 f16), stored row-major [M][ld] in memory.
// lane m = (lane&15); K values (lane>>4)*16 + 0..15 contiguous.
// Two 16B loads per lane.
__device__ inline v16h load_a_rowmajor(const _Float16* __restrict__ A, int ld,
                                       int mbase, int kc) {
    int lane = threadIdx.x & 31;
    const _Float16* p = A + (size_t)(mbase + (lane & 15)) * ld + kc + ((lane >> 4) << 4);
    v8h lo = *(const v8h*)p;
    v8h hi = *(const v8h*)(p + 8);
    return make_frag(lo, hi);
}

// B matrix (KxN, 32x16 f16), stored as [n][ld] (per-column contiguous K).
// lane column n = nbase + (lane&15); K values (lane>>4)*16 + 0..15 contiguous.
__device__ inline v16h load_b_colrows(const _Float16* __restrict__ Bt, int ld,
                                      int nbase, int kc) {
    int lane = threadIdx.x & 31;
    const _Float16* p = Bt + (size_t)(nbase + (lane & 15)) * ld + kc + ((lane >> 4) << 4);
    v8h lo = *(const v8h*)p;
    v8h hi = *(const v8h*)(p + 8);
    return make_frag(lo, hi);
}

// ---------------------------------------------------------------------------
// Prep kernels
// ---------------------------------------------------------------------------
__global__ void cvt_f16_kernel(const float* __restrict__ s, _Float16* __restrict__ d, int n) {
    int i = blockIdx.x * 256 + threadIdx.x;
    if (i < n) d[i] = (_Float16)s[i];
}

__global__ void bnfold_kernel(const float* __restrict__ cb, const float* __restrict__ g,
                              const float* __restrict__ bb, const float* __restrict__ m,
                              const float* __restrict__ v, float* __restrict__ scale,
                              float* __restrict__ bias) {
    int c = threadIdx.x;  // 32
    float s = g[c] * rsqrtf(v[c] + 1e-5f);
    scale[c] = s;
    bias[c] = s * (cb[c] - m[c]) + bb[c];
}

// A-stack: [b][m][64] f16, rows 0-31 = 2*mk, rows 32-63 = -mk^2.  Also value->f16.
__global__ void prep_attn_a_kernel(const float* __restrict__ proj, const float* __restrict__ value,
                                   _Float16* __restrict__ Ast, _Float16* __restrict__ mvh) {
    int b = blockIdx.x, m = threadIdx.x;
    for (int c = 0; c < 32; ++c) {
        float mk = proj[((size_t)(b * 8) * 32 + c) * 256 + m];   // frame 0
        Ast[((size_t)b * 256 + m) * 64 + c]      = (_Float16)(2.f * mk);
        Ast[((size_t)b * 256 + m) * 64 + 32 + c] = (_Float16)(-mk * mk);
        float mv = value[((size_t)b * 32 + c) * 256 + m];
        mvh[(size_t)b * 32 * 256 + (size_t)c * 256 + m] = (_Float16)mv;
    }
}

// B-stack: [bt][q][64] f16, rows 0-31 = qk*qe, rows 32-63 = qe.  b_sq per column.
__global__ void prep_attn_b_kernel(const float* __restrict__ proj, const float* __restrict__ qsel,
                                   _Float16* __restrict__ Bst, float* __restrict__ bsq) {
    int bt = blockIdx.x, q = threadIdx.x;
    int b = bt / 7, t = bt % 7 + 1;
    float s = 0.f;
    for (int c = 0; c < 32; ++c) {
        size_t si = ((size_t)(b * 8 + t) * 32 + c) * 256 + q;
        float qk = proj[si], qe = qsel[si];
        Bst[((size_t)bt * 256 + q) * 64 + c]      = (_Float16)(qk * qe);
        Bst[((size_t)bt * 256 + q) * 64 + 32 + c] = (_Float16)qe;
        s += qe * qk * qk;
    }
    bsq[bt * 256 + q] = s;
}

// ---------------------------------------------------------------------------
// Attention GEMMs (WMMA)
// ---------------------------------------------------------------------------
// sim[m,q] = (2ab - a_sq - b_sq[q]) * ms[m] / sqrt(C); store transposed [q][m].
__global__ __launch_bounds__(256) void sim_gemm_kernel(
    const _Float16* __restrict__ Ast, const _Float16* __restrict__ Bst,
    const float* __restrict__ bsq, const float* __restrict__ kscale,
    float* __restrict__ simT) {
    int bt = blockIdx.x, b = bt / 7;
    const _Float16* A  = Ast + (size_t)b * 256 * 64;
    const _Float16* Bt = Bst + (size_t)bt * 256 * 64;
    const float* bs = bsq + bt * 256;
    const float* ms = kscale + (size_t)b * 8 * 256;  // key_scale[b][0][0][*]
    float* out = simT + (size_t)bt * 65536;
    int lane = threadIdx.x & 31, wave = threadIdx.x >> 5;
    const float isc = 0.17677669529663687f;  // 1/sqrt(32)
    v16h a00 = load_a_rowmajor(A, 64, wave * 32, 0);
    v16h a01 = load_a_rowmajor(A, 64, wave * 32, 32);
    v16h a10 = load_a_rowmajor(A, 64, wave * 32 + 16, 0);
    v16h a11 = load_a_rowmajor(A, 64, wave * 32 + 16, 32);
    for (int nt = 0; nt < 16; ++nt) {
        v16h b0 = load_b_colrows(Bt, 64, nt * 16, 0);
        v16h b1 = load_b_colrows(Bt, 64, nt * 16, 32);
        v8f c0 = {}; c0 = WMMA_F16(a00, b0, c0); c0 = WMMA_F16(a01, b1, c0);
        v8f c1 = {}; c1 = WMMA_F16(a10, b0, c1); c1 = WMMA_F16(a11, b1, c1);
        int q = nt * 16 + (lane & 15);
        float bq = bs[q];
#pragma unroll
        for (int r = 0; r < 8; ++r) {
            int m0 = wave * 32 + r + ((lane >> 4) << 3);
            out[q * 256 + m0]      = (c0[r] - bq) * ms[m0] * isc;
            out[q * 256 + m0 + 16] = (c1[r] - bq) * ms[m0 + 16] * isc;
        }
    }
}

__global__ __launch_bounds__(256) void softmax_kernel(const float* __restrict__ simT,
                                                      _Float16* __restrict__ affT) {
    int bt = blockIdx.x, q = threadIdx.x;
    const float* row = simT + (size_t)bt * 65536 + (size_t)q * 256;
    float mx = -1e30f;
    for (int m = 0; m < 256; ++m) mx = fmaxf(mx, row[m]);
    float s = 0.f;
    for (int m = 0; m < 256; ++m) s += __expf(row[m] - mx);
    float inv = 1.f / s;
    _Float16* o = affT + (size_t)bt * 65536 + (size_t)q * 256;
    for (int m = 0; m < 256; ++m) o[m] = (_Float16)(__expf(row[m] - mx) * inv);
}

// mem[v,q] = sum_m mv[v,m]*aff[m,q]; epilogue writes channels 32..63 of concat.
__global__ __launch_bounds__(256) void mem_gemm_kernel(const _Float16* __restrict__ mvh,
                                                       const _Float16* __restrict__ affT,
                                                       _Float16* __restrict__ big) {
    int bt = blockIdx.x, b = bt / 7;
    const _Float16* A  = mvh + (size_t)b * 32 * 256;
    const _Float16* Bt = affT + (size_t)bt * 65536;
    int lane = threadIdx.x & 31, wave = threadIdx.x >> 5;
    v8f acc[2][2] = {};
    for (int kc = 0; kc < 256; kc += 32) {
        v16h a0 = load_a_rowmajor(A, 256, 0, kc);
        v16h a1 = load_a_rowmajor(A, 256, 16, kc);
        v16h b0 = load_b_colrows(Bt, 256, wave * 32, kc);
        v16h b1 = load_b_colrows(Bt, 256, wave * 32 + 16, kc);
        acc[0][0] = WMMA_F16(a0, b0, acc[0][0]);
        acc[0][1] = WMMA_F16(a0, b1, acc[0][1]);
        acc[1][0] = WMMA_F16(a1, b0, acc[1][0]);
        acc[1][1] = WMMA_F16(a1, b1, acc[1][1]);
    }
    _Float16* out = big + (size_t)bt * 64 * 256;
#pragma unroll
    for (int mt = 0; mt < 2; ++mt)
#pragma unroll
        for (int nt = 0; nt < 2; ++nt) {
            int q = wave * 32 + nt * 16 + (lane & 15);
#pragma unroll
            for (int r = 0; r < 8; ++r) {
                int v = mt * 16 + r + ((lane >> 4) << 3);
                out[(size_t)(32 + v) * 256 + q] = (_Float16)acc[mt][nt][r];
            }
        }
}

// ---------------------------------------------------------------------------
// Implicit-GEMM conv3x3 (pad 1), M=32, K=Cin*9, N=256 spatial per block.
// Fused BN (scale/bias) + optional ReLU epilogue; f16 in/out, f32 accumulate.
// ---------------------------------------------------------------------------
__global__ __launch_bounds__(256) void conv3x3_wmma_kernel(
    const _Float16* __restrict__ in, const _Float16* __restrict__ wf,
    const float* __restrict__ scale, const float* __restrict__ bias,
    _Float16* __restrict__ out, int H, int W, int Cin, int relu, int tilesPerImg) {
    __shared__ __align__(16) _Float16 Bsh[256 * 40];  // K-slice, stride-40 pad
    const int HW = H * W;
    const int K = Cin * 9;
    const int img = blockIdx.x / tilesPerImg;
    const int tile = blockIdx.x % tilesPerImg;
    const int pos0 = tile * 256;
    const int tid = threadIdx.x;
    const int lane = tid & 31, wave = tid >> 5;
    const _Float16* inImg = in + (size_t)img * Cin * HW;

    const int pcol = pos0 + tid;
    const int cy = pcol / W, cx = pcol - (pcol / W) * W;

    v8f acc[2][2] = {};
    for (int kc = 0; kc < K; kc += 32) {
        // im2col gather of the 32xK slice for this block's 256 columns
        for (int kk = 0; kk < 32; ++kk) {
            int k = kc + kk;
            int ci = k / 9, r = k - ci * 9;
            int ky = r / 3 - 1, kx = r - (r / 3) * 3 - 1;
            int yy = cy + ky, xx = cx + kx;
            _Float16 v = (_Float16)0.f;
            if (yy >= 0 && yy < H && xx >= 0 && xx < W)
                v = inImg[(size_t)ci * HW + yy * W + xx];
            Bsh[tid * 40 + kk] = v;
        }
        __syncthreads();
        v16h a0 = load_a_rowmajor(wf, K, 0, kc);
        v16h a1 = load_a_rowmajor(wf, K, 16, kc);
#pragma unroll
        for (int nt = 0; nt < 2; ++nt) {
            int n = wave * 32 + nt * 16 + (lane & 15);
            const _Float16* p = &Bsh[n * 40 + ((lane >> 4) << 4)];
            v8h lo = *(const v8h*)p;
            v8h hi = *(const v8h*)(p + 8);
            v16h bf = make_frag(lo, hi);
            acc[0][nt] = WMMA_F16(a0, bf, acc[0][nt]);
            acc[1][nt] = WMMA_F16(a1, bf, acc[1][nt]);
        }
        __syncthreads();
    }
#pragma unroll
    for (int mt = 0; mt < 2; ++mt)
#pragma unroll
        for (int nt = 0; nt < 2; ++nt) {
            int n = pos0 + wave * 32 + nt * 16 + (lane & 15);
#pragma unroll
            for (int r = 0; r < 8; ++r) {
                int m = mt * 16 + r + ((lane >> 4) << 3);
                float v = scale[m] * acc[mt][nt][r] + bias[m];
                if (relu) v = fmaxf(v, 0.f);
                out[(size_t)(img * 32 + m) * HW + n] = (_Float16)v;
            }
        }
}

// ---------------------------------------------------------------------------
// CBAM (channel MLP attention + 7x7 spatial attention), one block per image.
// ---------------------------------------------------------------------------
__global__ __launch_bounds__(256) void cbam_kernel(
    const _Float16* __restrict__ x, _Float16* __restrict__ y,
    const float* __restrict__ ca_w1, const float* __restrict__ ca_b1,
    const float* __restrict__ ca_w2, const float* __restrict__ ca_b2,
    const float* __restrict__ sa_w, const float* __restrict__ sa_b) {
    __shared__ float s_avg[32], s_mx[32], s_h[8], s_ca[32];
    __shared__ float sp0[256], sp1[256];
    __shared__ _Float16 xsc[32 * 256];
    int img = blockIdx.x, tid = threadIdx.x;
    const _Float16* xi = x + (size_t)img * 32 * 256;
    if (tid < 32) {
        float s = 0.f, m = -1e30f;
        for (int p = 0; p < 256; ++p) { float v = (float)xi[tid * 256 + p]; s += v; m = fmaxf(m, v); }
        s_avg[tid] = s * (1.f / 256.f); s_mx[tid] = m;
    }
    __syncthreads();
    if (tid < 8) {
        float ha = ca_b1[tid], hm = ca_b1[tid];
        for (int c = 0; c < 32; ++c) { ha += ca_w1[tid * 32 + c] * s_avg[c]; hm += ca_w1[tid * 32 + c] * s_mx[c]; }
        s_h[tid] = fmaxf(ha, 0.f) + fmaxf(hm, 0.f);
    }
    __syncthreads();
    if (tid < 32) {
        float o = 2.f * ca_b2[tid];
        for (int h = 0; h < 8; ++h) o += ca_w2[tid * 8 + h] * s_h[h];
        s_ca[tid] = 1.f / (1.f + __expf(-o));
    }
    __syncthreads();
    {   // apply channel attention; build spatial mean/max maps
        int p = tid; float s = 0.f, m = -1e30f;
        for (int c = 0; c < 32; ++c) {
            float v = (float)xi[c * 256 + p] * s_ca[c];
            xsc[c * 256 + p] = (_Float16)v; s += v; m = fmaxf(m, v);
        }
        sp0[p] = s * (1.f / 32.f); sp1[p] = m;
    }
    __syncthreads();
    {   // 7x7 spatial conv (2ch -> 1) + sigmoid, multiply through
        int p = tid, py = p >> 4, px = p & 15;
        float a = sa_b[0];
        for (int ky = 0; ky < 7; ++ky) {
            int yy = py + ky - 3; if (yy < 0 || yy >= 16) continue;
            for (int kx = 0; kx < 7; ++kx) {
                int xx = px + kx - 3; if (xx < 0 || xx >= 16) continue;
                a += sa_w[ky * 7 + kx] * sp0[yy * 16 + xx] + sa_w[49 + ky * 7 + kx] * sp1[yy * 16 + xx];
            }
        }
        float sg = 1.f / (1.f + __expf(-a));
        _Float16* yo = y + (size_t)img * 32 * 256;
        for (int c = 0; c < 32; ++c) yo[c * 256 + p] = (_Float16)((float)xsc[c * 256 + p] * sg);
    }
}

// ---------------------------------------------------------------------------
// Small elementwise / reduction kernels
// ---------------------------------------------------------------------------
__global__ void build_concat32_kernel(const float* __restrict__ f32, _Float16* __restrict__ big) {
    int idx = blockIdx.x * 256 + threadIdx.x;  // 56*32*256
    int p = idx & 255, rest = idx >> 8, c = rest & 31, img = rest >> 5;
    int b = img / 7, t = img % 7 + 1;
    float v = f32[((size_t)(b * 8 + t) * 32 + c) * 256 + p];
    big[((size_t)img * 64 + c) * 256 + p] = (_Float16)v;
}

__global__ void concat_sig_f32_kernel(const float* __restrict__ f, _Float16* __restrict__ big, int HW) {
    int idx = blockIdx.x * 256 + threadIdx.x;  // 56*32*HW
    int p = idx % HW, rest = idx / HW, c = rest & 31, img = rest >> 5;
    int b = img / 7, t = img % 7 + 1;
    float v = f[((size_t)(b * 8 + t) * 32 + c) * HW + p];
    big[((size_t)img * 64 + c) * HW + p] = (_Float16)v;
    big[((size_t)img * 64 + 32 + c) * HW + p] = (_Float16)(1.f - 1.f / (1.f + __expf(-v)));
}

__global__ void concat_sig_f16_kernel(const _Float16* __restrict__ f, _Float16* __restrict__ big, int HW) {
    int idx = blockIdx.x * 256 + threadIdx.x;
    int p = idx % HW, rest = idx / HW, c = rest & 31, img = rest >> 5;
    float v = (float)f[((size_t)img * 32 + c) * HW + p];
    big[((size_t)img * 64 + c) * HW + p] = (_Float16)v;
    big[((size_t)img * 64 + 32 + c) * HW + p] = (_Float16)(1.f - 1.f / (1.f + __expf(-v)));
}

__global__ void conv1x1_kernel(const _Float16* __restrict__ in, const float* __restrict__ w,
                               const float* __restrict__ b, float* __restrict__ out, int HW) {
    int idx = blockIdx.x * 256 + threadIdx.x;
    int img = idx / HW, p = idx % HW;
    float a = b[0];
    const _Float16* ip = in + (size_t)img * 32 * HW + p;
    for (int c = 0; c < 32; ++c) a += w[c] * (float)ip[(size_t)c * HW];
    out[(size_t)img * HW + p] = a;
}

__global__ void upsample_kernel(const float* __restrict__ in, float* __restrict__ out, int H, int W) {
    int idx = blockIdx.x * 256 + threadIdx.x;  // 56*512*512
    int img = idx >> 18, r = idx & 262143;
    int oy = r >> 9, ox = r & 511;
    float sy = (float)oy * (float)(H - 1) * (1.f / 511.f);
    float sx = (float)ox * (float)(W - 1) * (1.f / 511.f);
    int y0 = (int)floorf(sy); y0 = y0 < 0 ? 0 : (y0 > H - 2 ? H - 2 : y0);
    int x0 = (int)floorf(sx); x0 = x0 < 0 ? 0 : (x0 > W - 2 ? W - 2 : x0);
    float wy = sy - (float)y0, wx = sx - (float)x0;
    const float* ip = in + (size_t)img * H * W;
    float top = ip[y0 * W + x0] * (1.f - wx) + ip[y0 * W + x0 + 1] * wx;
    float bot = ip[(y0 + 1) * W + x0] * (1.f - wx) + ip[(y0 + 1) * W + x0 + 1] * wx;
    out[idx] = top * (1.f - wy) + bot * wy;
}

// ---------------------------------------------------------------------------
// Host launch
// ---------------------------------------------------------------------------
extern "C" void kernel_launch(void* const* d_in, const int* in_sizes, int n_in,
                              void* d_out, int out_size, void* d_ws, size_t ws_size,
                              hipStream_t stream) {
    (void)in_sizes; (void)out_size; (void)ws_size;
    // Param leaves in dict insertion order (per _make_params)
    static const int NP = 96;
    static const int PSIZES[NP] = {
        18432, 32, 32, 32, 32, 32,          // mf0  w,b,g,bb,m,v
        9216, 32, 32, 32, 32, 32,           // mf1
        256, 8, 256, 32,                    // ca_w1, ca_b1, ca_w2, ca_b2
        98, 1,                              // sa_w, sa_b
        18432, 32, 32, 32, 32, 32,          // d32_0
        9216, 32, 32, 32, 32, 32,           // d32_1
        9216, 32, 32, 32, 32, 32,           // d32_2
        9216, 32, 32, 32, 32, 32,           // d32_3
        32, 1,                              // d32_4 w,b
        18432, 32, 32, 32, 32, 32,          // d16_0
        9216, 32, 32, 32, 32, 32,
        9216, 32, 32, 32, 32, 32,
        9216, 32, 32, 32, 32, 32,
        32, 1,                              // d16_4
        18432, 32, 32, 32, 32, 32,          // d8_0
        9216, 32, 32, 32, 32, 32,
        9216, 32, 32, 32, 32, 32,
        9216, 32, 32, 32, 32, 32,
        32, 1                               // d8_4
    };
    const float* P[NP];
    if (n_in >= 7 + NP) {
        for (int i = 0; i < NP; ++i) P[i] = (const float*)d_in[7 + i];
    } else {
        const float* base = (const float*)d_in[7];
        size_t off = 0;
        for (int i = 0; i < NP; ++i) { P[i] = base + off; off += (size_t)PSIZES[i]; }
    }

    const float* f8in  = (const float*)d_in[0];
    const float* f16in = (const float*)d_in[1];
    const float* f32in = (const float*)d_in[2];
    const float* proj  = (const float*)d_in[3];
    const float* ksc   = (const float*)d_in[4];
    const float* qsel  = (const float*)d_in[5];
    const float* value = (const float*)d_in[6];
    float* out = (float*)d_out;

    // Workspace bump allocator (everything stays inside the 192MB L2)
    char* wsb = (char*)d_ws;
    size_t off = 0;
    auto alloc = [&](size_t bytes) -> char* {
        char* p = wsb + off;
        off = (off + bytes + 255) & ~(size_t)255;
        return p;
    };

    // 14 conv3x3 layers: index -> param base, K
    static const int LBASE[14] = {0, 6, 18, 24, 30, 36, 44, 50, 56, 62, 70, 76, 82, 88};
    static const int LK[14]    = {576, 288, 576, 288, 288, 288, 576, 288, 288, 288, 576, 288, 288, 288};
    _Float16* wf16[14];
    for (int i = 0; i < 14; ++i) wf16[i] = (_Float16*)alloc((size_t)LK[i] * 32 * 2);
    float* scl = (float*)alloc(14 * 32 * 4);
    float* bia = (float*)alloc(14 * 32 * 4);
    _Float16* Ast = (_Float16*)alloc((size_t)8 * 256 * 64 * 2);
    _Float16* Bst = (_Float16*)alloc((size_t)56 * 256 * 64 * 2);
    float*    bsq = (float*)alloc((size_t)56 * 256 * 4);
    _Float16* mvh = (_Float16*)alloc((size_t)8 * 32 * 256 * 2);
    float*    simT = (float*)alloc((size_t)56 * 65536 * 4);
    _Float16* affT = (_Float16*)alloc((size_t)56 * 65536 * 2);
    _Float16* big  = (_Float16*)alloc((size_t)56 * 64 * 4096 * 2);  // 64ch concat (reused)
    _Float16* pA   = (_Float16*)alloc((size_t)56 * 32 * 4096 * 2);  // ping
    _Float16* pB   = (_Float16*)alloc((size_t)56 * 32 * 4096 * 2);  // pong
    float* m32 = (float*)alloc((size_t)56 * 256 * 4);
    float* m16 = (float*)alloc((size_t)56 * 1024 * 4);
    float* m8  = (float*)alloc((size_t)56 * 4096 * 4);

    // Weight conversion + BN folding
    for (int i = 0; i < 14; ++i) {
        int n = LK[i] * 32;
        cvt_f16_kernel<<<(n + 255) / 256, 256, 0, stream>>>(P[LBASE[i]], wf16[i], n);
        bnfold_kernel<<<1, 32, 0, stream>>>(P[LBASE[i] + 1], P[LBASE[i] + 2], P[LBASE[i] + 3],
                                            P[LBASE[i] + 4], P[LBASE[i] + 5],
                                            scl + i * 32, bia + i * 32);
    }

    // Attention
    prep_attn_a_kernel<<<8, 256, 0, stream>>>(proj, value, Ast, mvh);
    prep_attn_b_kernel<<<56, 256, 0, stream>>>(proj, qsel, Bst, bsq);
    sim_gemm_kernel<<<56, 256, 0, stream>>>(Ast, Bst, bsq, ksc, simT);
    softmax_kernel<<<56, 256, 0, stream>>>(simT, affT);
    mem_gemm_kernel<<<56, 256, 0, stream>>>(mvh, affT, big);      // ch 32..63
    build_concat32_kernel<<<(56 * 32 * 256) / 256, 256, 0, stream>>>(f32in, big);  // ch 0..31

    // Fuse head: mf0 -> CBAM -> mf1
    conv3x3_wmma_kernel<<<56, 256, 0, stream>>>(big, wf16[0], scl + 0 * 32, bia + 0 * 32, pA, 16, 16, 64, 1, 1);
    cbam_kernel<<<56, 256, 0, stream>>>(pA, pB, P[12], P[13], P[14], P[15], P[16], P[17]);
    conv3x3_wmma_kernel<<<56, 256, 0, stream>>>(pB, wf16[1], scl + 1 * 32, bia + 1 * 32, pA, 16, 16, 32, 1, 1);

    // d32 decoder
    concat_sig_f16_kernel<<<(56 * 32 * 256) / 256, 256, 0, stream>>>(pA, big, 256);
    conv3x3_wmma_kernel<<<56, 256, 0, stream>>>(big, wf16[2], scl + 2 * 32, bia + 2 * 32, pA, 16, 16, 64, 1, 1);
    conv3x3_wmma_kernel<<<56, 256, 0, stream>>>(pA, wf16[3], scl + 3 * 32, bia + 3 * 32, pB, 16, 16, 32, 1, 1);
    conv3x3_wmma_kernel<<<56, 256, 0, stream>>>(pB, wf16[4], scl + 4 * 32, bia + 4 * 32, pA, 16, 16, 32, 1, 1);
    conv3x3_wmma_kernel<<<56, 256, 0, stream>>>(pA, wf16[5], scl + 5 * 32, bia + 5 * 32, pB, 16, 16, 32, 1, 1);
    conv1x1_kernel<<<56, 256, 0, stream>>>(pB, P[42], P[43], m32, 256);

    // d16 decoder
    concat_sig_f32_kernel<<<(56 * 32 * 1024) / 256, 256, 0, stream>>>(f16in, big, 1024);
    conv3x3_wmma_kernel<<<56 * 4, 256, 0, stream>>>(big, wf16[6], scl + 6 * 32, bia + 6 * 32, pA, 32, 32, 64, 1, 4);
    conv3x3_wmma_kernel<<<56 * 4, 256, 0, stream>>>(pA, wf16[7], scl + 7 * 32, bia + 7 * 32, pB, 32, 32, 32, 1, 4);
    conv3x3_wmma_kernel<<<56 * 4, 256, 0, stream>>>(pB, wf16[8], scl + 8 * 32, bia + 8 * 32, pA, 32, 32, 32, 1, 4);
    conv3x3_wmma_kernel<<<56 * 4, 256, 0, stream>>>(pA, wf16[9], scl + 9 * 32, bia + 9 * 32, pB, 32, 32, 32, 1, 4);
    conv1x1_kernel<<<(56 * 1024) / 256, 256, 0, stream>>>(pB, P[68], P[69], m16, 1024);

    // d8 decoder
    concat_sig_f32_kernel<<<(56 * 32 * 4096) / 256, 256, 0, stream>>>(f8in, big, 4096);
    conv3x3_wmma_kernel<<<56 * 16, 256, 0, stream>>>(big, wf16[10], scl + 10 * 32, bia + 10 * 32, pA, 64, 64, 64, 1, 16);
    conv3x3_wmma_kernel<<<56 * 16, 256, 0, stream>>>(pA, wf16[11], scl + 11 * 32, bia + 11 * 32, pB, 64, 64, 32, 1, 16);
    conv3x3_wmma_kernel<<<56 * 16, 256, 0, stream>>>(pB, wf16[12], scl + 12 * 32, bia + 12 * 32, pA, 64, 64, 32, 1, 16);
    conv3x3_wmma_kernel<<<56 * 16, 256, 0, stream>>>(pA, wf16[13], scl + 13 * 32, bia + 13 * 32, pB, 64, 64, 32, 1, 16);
    conv1x1_kernel<<<(56 * 4096) / 256, 256, 0, stream>>>(pB, P[94], P[95], m8, 4096);

    // Bilinear upsample (align_corners) to 512x512; output order (m8, m16, m32)
    const size_t sect = (size_t)56 * 262144;
    upsample_kernel<<<57344, 256, 0, stream>>>(m8, out, 64, 64);
    upsample_kernel<<<57344, 256, 0, stream>>>(m16, out + sect, 32, 32);
    upsample_kernel<<<57344, 256, 0, stream>>>(m32, out + 2 * sect, 16, 16);
}
